// BinaryBiaffine2_13692355739775
// MI455X (gfx1250) — matrix operand order repaired and checked by reading, not compiled, not run on hardware
//
#include <hip/hip_runtime.h>

// ---------------------------------------------------------------------------
// BinaryBiaffine2 for MI455X (gfx1250, wave32, WMMA).
//
// Stage 1 (matrix-core bound): scores[row][4] = {dep0,dep1,head0,head1}
//   via split-bf16 (hi/lo) WMMA GEMM, fp32 accumulate:  a*w ~= ah*wh + ah*wl + al*wh
// Stage 2 (HBM bound): out[b,i,j,c] = dep[b,i,c] + head[b,j,c] + bc[c]  (128 MB)
// ---------------------------------------------------------------------------

typedef __attribute__((ext_vector_type(16))) __bf16 v16bf;
typedef __attribute__((ext_vector_type(8)))  float  v8f;

#define ROWS   16384   // B*L
#define DDIM   1024    // K
#define KCHUNK 32      // K per WMMA
#define NKC    (DDIM / KCHUNK)        // 32 k-chunks
#define NPAD   512     // padded columns per branch (500 -> 512)
#define NTILES 64      // (2*NPAD)/16 column tiles
#define MLPD   500

// ---------------------------------------------------------------------------
// Prep: split fp32 weights into bf16 hi/lo, packed in a contiguous-K wave32
// B-fragment order (A uses the same) for V_WMMA_F32_16X16X32_BF16 (32x16 KxN tile):
//   lanes 0-15: N = lane,    K = 0..15 ;  lanes 16-31: N = lane-16, K = 16..31
//   per lane: 16 bf16, K ascending, 2 per VGPR.
// Wf block for (nt, kc): 1024 bf16 = [hi: lane*16 .. | lo: 512 + lane*16 ..]
// ---------------------------------------------------------------------------
__global__ __launch_bounds__(256) void prep_w_kernel(
    const float* __restrict__ Wdep, const float* __restrict__ Whead,
    __bf16* __restrict__ Wf)
{
  int t    = blockIdx.x * 256 + threadIdx.x;     // 0 .. 64*32*32-1
  int lane = t & 31;
  int kc   = (t >> 5) & 31;
  int nt   = t >> 10;                            // 0..63
  int koff = (lane < 16) ? 0 : 16;
  int ng   = nt * 16 + (lane & 15);              // padded global column
  int cib  = ng & (NPAD - 1);                    // column in branch
  bool head  = ng >= NPAD;
  bool valid = cib < MLPD;
  const float* W = head ? Whead : Wdep;

  __bf16 hi[16], lo[16];
#pragma unroll
  for (int i = 0; i < 16; ++i) {
    int k   = kc * KCHUNK + koff + i;
    float v = valid ? W[(size_t)k * MLPD + cib] : 0.0f;
    __bf16 h = (__bf16)v;
    hi[i] = h;
    lo[i] = (__bf16)(v - (float)h);
  }
  __bf16* dst = Wf + (size_t)(nt * NKC + kc) * 1024 + lane * 16;
#pragma unroll
  for (int i = 0; i < 16; ++i) { dst[i] = hi[i]; dst[512 + i] = lo[i]; }
}

// ---------------------------------------------------------------------------
// Stage 1: per-wave 16-row M-tile x 8 N-tiles, K-outer loop.
// A fragment (16x32 bf16): lane m = lane&15 (row), K segments
//   {s..s+7, s+8..s+15} with s = (lane<16 ? 0 : 16) -> two contiguous
//   32B fp32 reads per segment, converted to bf16 hi/lo in-register.
// Epilogue: bias + LeakyReLU + (x Wc) projection fused, xor-shuffle
// reduction over the 16-lane halves, deterministic per-n-group partials.
// ---------------------------------------------------------------------------
__global__ __launch_bounds__(256) void gemm_scores_kernel(
    const float* __restrict__ hidden,
    const float* __restrict__ b_dep, const float* __restrict__ b_head,
    const float* __restrict__ Wc,
    const __bf16* __restrict__ Wf,
    float* __restrict__ part)                    // [8][ROWS][2]
{
  const int lane = threadIdx.x & 31;
  const int wave = threadIdx.x >> 5;
  const int ng   = blockIdx.y;                   // n-group 0..7 (0-3 dep, 4-7 head)
  const int row0 = blockIdx.x * 128 + wave * 16;
  const int m    = lane & 15;
  const int sA   = (lane < 16) ? 0 : 16;
  const float* arow = hidden + (size_t)(row0 + m) * DDIM;

  v8f acc[8];
#pragma unroll
  for (int i = 0; i < 8; ++i) acc[i] = (v8f){0.f,0.f,0.f,0.f,0.f,0.f,0.f,0.f};

  for (int kc = 0; kc < NKC; ++kc) {
    const int kb = kc * KCHUNK;
    // A: 16 fp32 per lane (two 32B contiguous segments)
    const float4* pa = (const float4*)(arow + kb + sA);
    const float4* pb = (const float4*)(arow + kb + sA + 8);
    float4 f0 = pa[0], f1 = pa[1], f2 = pb[0], f3 = pb[1];
    float av[16] = { f0.x,f0.y,f0.z,f0.w, f1.x,f1.y,f1.z,f1.w,
                     f2.x,f2.y,f2.z,f2.w, f3.x,f3.y,f3.z,f3.w };
    v16bf ahi, alo;
#pragma unroll
    for (int j = 0; j < 16; ++j) {
      __bf16 h = (__bf16)av[j];
      ahi[j] = h;
      alo[j] = (__bf16)(av[j] - (float)h);
    }
#pragma unroll
    for (int nt = 0; nt < 8; ++nt) {
      const __bf16* bp = Wf + (size_t)((ng * 8 + nt) * NKC + kc) * 1024 + lane * 16;
      v16bf bhi = *(const v16bf*)bp;
      v16bf blo = *(const v16bf*)(bp + 512);
      // small terms first, then dominant term (fp32 accumulate throughout)
      acc[nt] = __builtin_amdgcn_wmma_f32_16x16x32_bf16(false, alo, false, bhi,
                                                        (short)0, acc[nt], false, false);
      acc[nt] = __builtin_amdgcn_wmma_f32_16x16x32_bf16(false, ahi, false, blo,
                                                        (short)0, acc[nt], false, false);
      acc[nt] = __builtin_amdgcn_wmma_f32_16x16x32_bf16(false, ahi, false, bhi,
                                                        (short)0, acc[nt], false, false);
    }
  }

  // Fused epilogue: bias + LeakyReLU + projection onto Wc columns.
  const bool is_head = (ng >= 4);
  float p0[8], p1[8];
#pragma unroll
  for (int r = 0; r < 8; ++r) { p0[r] = 0.f; p1[r] = 0.f; }

#pragma unroll
  for (int nt = 0; nt < 8; ++nt) {
    int ncol  = (ng * 8 + nt) * 16 + m;          // padded combined column
    int cib   = ncol & (NPAD - 1);
    bool valid = cib < MLPD;
    float bias = 0.f, w0 = 0.f, w1 = 0.f;
    if (valid) {
      bias = is_head ? b_head[cib] : b_dep[cib];
      int wr = (is_head ? MLPD : 0) + cib;       // Wc is [2*MLPD][2]
      w0 = Wc[wr * 2 + 0];
      w1 = Wc[wr * 2 + 1];
    }
#pragma unroll
    for (int r = 0; r < 8; ++r) {
      float x = acc[nt][r] + bias;
      float a = x > 0.f ? x : 0.01f * x;         // LeakyReLU(0.01)
      p0[r] += a * w0;
      p1[r] += a * w1;
    }
  }

  // Reduce across the 16 lanes of each half (C layout: lanes<16 -> rows r,
  // lanes>=16 -> rows r+8). xor masks 1,2,4,8 stay within each half.
#pragma unroll
  for (int r = 0; r < 8; ++r) {
#pragma unroll
    for (int off = 1; off < 16; off <<= 1) {
      p0[r] += __shfl_xor(p0[r], off, 32);
      p1[r] += __shfl_xor(p1[r], off, 32);
    }
  }

  if (m == 0) {                                  // lanes 0 and 16 write
    int half = lane >> 4;
#pragma unroll
    for (int r = 0; r < 8; ++r) {
      size_t rr = (size_t)row0 + half * 8 + r;
      part[((size_t)ng * ROWS + rr) * 2 + 0] = p0[r];
      part[((size_t)ng * ROWS + rr) * 2 + 1] = p1[r];
    }
  }
}

// ---------------------------------------------------------------------------
// Stage 1.5: fold the 8 n-group partials into scores[row][4] = {d0,d1,h0,h1}
// ---------------------------------------------------------------------------
__global__ __launch_bounds__(256) void reduce_scores_kernel(
    const float* __restrict__ part, float* __restrict__ scores)
{
  int r = blockIdx.x * 256 + threadIdx.x;        // 0..ROWS-1
  float d0 = 0.f, d1 = 0.f, h0 = 0.f, h1 = 0.f;
#pragma unroll
  for (int g = 0; g < 4; ++g) {
    d0 += part[((size_t)g * ROWS + r) * 2 + 0];
    d1 += part[((size_t)g * ROWS + r) * 2 + 1];
  }
#pragma unroll
  for (int g = 4; g < 8; ++g) {
    h0 += part[((size_t)g * ROWS + r) * 2 + 0];
    h1 += part[((size_t)g * ROWS + r) * 2 + 1];
  }
  float4* s = (float4*)(scores + (size_t)r * 4);
  *s = (float4){d0, d1, h0, h1};
}

// ---------------------------------------------------------------------------
// Stage 2: out[b,i,j,c] = dep[b,i,c] + head[b,j,c] + bc[c].  Pure bandwidth:
// one float4 store (two j's) per thread, 128 MB total.
// ---------------------------------------------------------------------------
__global__ __launch_bounds__(256) void broadcast_kernel(
    const float* __restrict__ scores, const float* __restrict__ bc,
    float* __restrict__ out)
{
  size_t t = (size_t)blockIdx.x * 256 + threadIdx.x;   // 0 .. 8M-1
  int jp = (int)(t & 511);                             // j pair
  int i  = (int)((t >> 9) & 1023);
  int b  = (int)(t >> 19);
  float bc0 = bc[0], bc1 = bc[1];
  const float* si = scores + ((size_t)b * 1024 + i) * 4;
  float d0 = si[0], d1 = si[1];
  const float* sj = scores + ((size_t)b * 1024 + jp * 2) * 4;
  float4 o = { d0 + sj[2] + bc0, d1 + sj[3] + bc1,
               d0 + sj[6] + bc0, d1 + sj[7] + bc1 };
  ((float4*)out)[t] = o;
}

// ---------------------------------------------------------------------------
extern "C" void kernel_launch(void* const* d_in, const int* in_sizes, int n_in,
                              void* d_out, int out_size, void* d_ws, size_t ws_size,
                              hipStream_t stream) {
  (void)in_sizes; (void)n_in; (void)out_size; (void)ws_size;
  const float* hidden = (const float*)d_in[0];
  const float* Wdep   = (const float*)d_in[1];
  const float* bdep   = (const float*)d_in[2];
  const float* Whead  = (const float*)d_in[3];
  const float* bhead  = (const float*)d_in[4];
  const float* Wc     = (const float*)d_in[5];
  const float* bc     = (const float*)d_in[6];
  float* out = (float*)d_out;

  char* ws = (char*)d_ws;
  float*  scores = (float*)ws;                         // 16384*4 f32 = 256 KB
  float*  part   = (float*)(ws + (256u << 10));        // 8*16384*2 f32 = 1 MB
  __bf16* Wf     = (__bf16*)(ws + (256u << 10) + (1u << 20));  // 4 MB packed frags

  prep_w_kernel<<<256, 256, 0, stream>>>(Wdep, Whead, Wf);

  dim3 g1(128, 8);
  gemm_scores_kernel<<<g1, 256, 0, stream>>>(hidden, bdep, bhead, Wc, Wf, part);

  reduce_scores_kernel<<<ROWS / 256, 256, 0, stream>>>(part, scores);

  broadcast_kernel<<<(16u * 1024u * 1024u * 2u / 4u) / 256u, 256, 0, stream>>>(
      scores, bc, out);
}
